// MultiHeadAttention_74955769250711
// MI455X (gfx1250) — matrix operand order repaired and checked by reading, not compiled, not run on hardware
//
#include <hip/hip_runtime.h>
#include <hip/hip_bf16.h>

// MI455X / gfx1250, wave32, WMMA bf16 path.
// B=4, L=1024, D=512, H=8, DK=64, M=1024 (fixed problem size).

typedef __attribute__((ext_vector_type(16))) __bf16 v16bf;
typedef __attribute__((ext_vector_type(8)))  float  v8f;

#define B_   4
#define L_   1024
#define D_   512
#define H_   8
#define DK_  64
#define NROW (B_*L_)   // 4096

struct U8x32 { uint4 lo; uint4 hi; };

static __device__ __forceinline__ v16bf pack_v16(uint4 lo, uint4 hi) {
  U8x32 t; t.lo = lo; t.hi = hi;
  return __builtin_bit_cast(v16bf, t);
}
static __device__ __forceinline__ unsigned short f2bf(float f) {
  unsigned int u = __builtin_bit_cast(unsigned int, f);
  u += 0x7fffu + ((u >> 16) & 1u);       // round-to-nearest-even
  return (unsigned short)(u >> 16);
}
static __device__ __forceinline__ float bf2f(unsigned short h) {
  unsigned int u = ((unsigned int)h) << 16;
  return __builtin_bit_cast(float, u);
}

// ---------------------------------------------------------------- converts
__global__ void k_f32_to_bf16(const float* __restrict__ in,
                              unsigned short* __restrict__ out, int n) {
  int i = blockIdx.x * 256 + threadIdx.x;
  if (i < n) out[i] = f2bf(in[i]);
}

// tp_diag[m,d] = rel_time[m,m,d] + rel_pitch[m,m,d]   (only the diagonal!)
__global__ void k_tp_diag(const float* __restrict__ rt,
                          const float* __restrict__ rp,
                          float* __restrict__ tp) {
  int i = blockIdx.x * 256 + threadIdx.x;   // 65536 total
  int m = i >> 6, d = i & 63;
  size_t src = ((size_t)m * L_ + m) * DK_ + d;
  tp[i] = rt[src] + rp[src];
}

// ---------------------------------------------------------------- projection
// Y = X @ W^T + bias ; X:[4096,512] bf16, W:[512,512] bf16 (row=out,col=in)
// Register-blocked: each wave computes a 32x64 macro-tile (2x4 WMMA tiles).
// mode 0: store bf16 head-split  [B,H,L,DK]           (Q)
// mode 1: same + tp_diag[pos,dk]                       (K')
// mode 2: store bf16 head-split transposed [B,H,DK,L]  (V^T)
// mode 3: store f32 flat [N,512]                       (final out-proj)
__global__ void __launch_bounds__(256)
k_proj(const unsigned short* __restrict__ X,
       const unsigned short* __restrict__ W,
       const float* __restrict__ bias,
       const float* __restrict__ tp,
       void* __restrict__ out, int mode) {
  int wid  = threadIdx.x >> 5;
  int lane = threadIdx.x & 31;
  int t  = blockIdx.x * 8 + wid;           // 1024 macro tiles: 128 row x 8 col
  int cm = t & 7, rm = t >> 3;
  int n0 = rm * 32, o0 = cm * 64;
  int l15 = lane & 15;
  int ka = (lane < 16) ? 0 : 16;           // A k-offset pattern
  int kb = (lane < 16) ? 0 : 16;           // B k-offset pattern

  const unsigned short* arow0 = X + (size_t)(n0 + l15) * D_;
  const unsigned short* arow1 = arow0 + (size_t)16 * D_;
  const unsigned short* brow  = W + (size_t)(o0 + l15) * D_;

  v8f acc[8] = {};
#pragma unroll 4
  for (int kc = 0; kc < D_; kc += 32) {
    v16bf a0 = pack_v16(*(const uint4*)(arow0 + kc + ka),
                        *(const uint4*)(arow0 + kc + ka + 8));
    v16bf a1 = pack_v16(*(const uint4*)(arow1 + kc + ka),
                        *(const uint4*)(arow1 + kc + ka + 8));
#pragma unroll
    for (int c = 0; c < 4; ++c) {
      const unsigned short* br = brow + (size_t)c * 16 * D_;
      v16bf bv = pack_v16(*(const uint4*)(br + kc + kb),
                          *(const uint4*)(br + kc + kb + 8));
      acc[2*c+0] = __builtin_amdgcn_wmma_f32_16x16x32_bf16(false, a0, false, bv,
                                      (short)0, acc[2*c+0], false, false);
      acc[2*c+1] = __builtin_amdgcn_wmma_f32_16x16x32_bf16(false, a1, false, bv,
                                      (short)0, acc[2*c+1], false, false);
    }
  }

  int hh = lane >> 4;
#pragma unroll
  for (int c = 0; c < 4; ++c) {
    int o = o0 + c * 16 + l15;
    float bval = bias[o];
    int h = o >> 6, dk = o & 63;
#pragma unroll
    for (int rt2 = 0; rt2 < 2; ++rt2) {
      v8f a = acc[2*c + rt2];
      for (int r = 0; r < 8; ++r) {
        int n = n0 + rt2 * 16 + r + hh * 8;
        float vv = a[r] + bval;
        int b = n >> 10, i = n & 1023;
        if (mode == 1) vv += tp[i * DK_ + dk];
        if (mode <= 1) {
          ((unsigned short*)out)[(((size_t)b * H_ + h) * L_ + i) * DK_ + dk] = f2bf(vv);
        } else if (mode == 2) {
          ((unsigned short*)out)[(((size_t)b * H_ + h) * DK_ + dk) * L_ + i] = f2bf(vv);
        } else {
          ((float*)out)[(size_t)n * D_ + o] = vv;
        }
      }
    }
  }
}

// ---------------------------------------------------------------- attention
// One block = (b, h, 16-row slab). Scores (bf16) + P (bf16 in-place) in LDS.
__global__ void __launch_bounds__(256)
k_attn(const unsigned short* __restrict__ Qh,
       const unsigned short* __restrict__ Kp,
       const unsigned short* __restrict__ Vt,
       const unsigned short* __restrict__ Rk,
       unsigned short* __restrict__ Oc) {
  __shared__ __align__(16) unsigned short sc[16 * 1024];  // scores -> P (bf16)
  __shared__ __align__(16) unsigned short qt[16 * 64];    // Q slab (bf16)
  __shared__ float red[256];
  __shared__ float opart[4 * 16 * 16];

  int tid  = threadIdx.x;
  int wid  = tid >> 5, lane = tid & 31;
  int ib = blockIdx.x & 63;
  int h  = (blockIdx.x >> 6) & 7;
  int b  =  blockIdx.x >> 9;
  int i0 = ib << 4;
  int bh = b * H_ + h;

  const unsigned short* Qb = Qh + (size_t)bh * L_ * DK_;
  const unsigned short* Kb = Kp + (size_t)bh * L_ * DK_;
  const unsigned short* Vb = Vt + (size_t)bh * DK_ * L_;
  const unsigned short* Rb = Rk + (size_t)h  * L_ * DK_;   // M == L

  { // stage Q slab (16x64 bf16 = 2KB) into LDS via CDNA5 async copy (ASYNCcnt)
    const uint4* src = (const uint4*)(Qb + (size_t)i0 * DK_);
    if (tid < 128) {
      unsigned lds_addr = (unsigned)(size_t)(&qt[0]) + (unsigned)tid * 16u;
      unsigned long long gaddr = (unsigned long long)(size_t)(src + tid);
      asm volatile("global_load_async_to_lds_b128 %0, %1, off"
                   :: "v"(lds_addr), "v"(gaddr) : "memory");
    }
    asm volatile("s_wait_asynccnt 0x0" ::: "memory");
  }
  __syncthreads();

  int l15 = lane & 15;
  int ka = (lane < 16) ? 0 : 16;
  int kb = (lane < 16) ? 0 : 16;
  int hh = lane >> 4;
  int nj16 = ib + 1;                        // causal: tiles with j <= i0+15

  // ---- pass 1a: S = Q @ K'^T  (tp-diag term already folded into K')
  for (int t = wid; t < nj16; t += 8) {
    int j0 = t << 4;
    if (t + 8 < nj16)
      __builtin_prefetch(Kb + (size_t)((t + 8) * 16 + l15) * DK_, 0, 1);
    const unsigned short* brow = Kb + (size_t)(j0 + l15) * DK_;
    const unsigned short* arow = qt + l15 * DK_;
    v8f acc = {};
    for (int kc = 0; kc < DK_; kc += 32) {
      v16bf av = pack_v16(*(const uint4*)(arow + kc + ka),
                          *(const uint4*)(arow + kc + ka + 8));
      v16bf bv = pack_v16(*(const uint4*)(brow + kc + kb),
                          *(const uint4*)(brow + kc + kb + 8));
      acc = __builtin_amdgcn_wmma_f32_16x16x32_bf16(false, av, false, bv,
                                                    (short)0, acc, false, false);
    }
    for (int r = 0; r < 8; ++r)
      sc[(r + hh * 8) * 1024 + j0 + l15] = f2bf(acc[r]);
  }
  __syncthreads();

  // ---- pass 1b: rel logits in (i,m) space; skew-scatter j = m-(L-1)+i
  int mstart = L_ - 16 - i0;
  for (int t = wid; t < nj16; t += 8) {
    int m0 = mstart + (t << 4);
    const unsigned short* brow = Rb + (size_t)(m0 + l15) * DK_;
    const unsigned short* arow = qt + l15 * DK_;
    v8f acc = {};
    for (int kc = 0; kc < DK_; kc += 32) {
      v16bf av = pack_v16(*(const uint4*)(arow + kc + ka),
                          *(const uint4*)(arow + kc + ka + 8));
      v16bf bv = pack_v16(*(const uint4*)(brow + kc + kb),
                          *(const uint4*)(brow + kc + kb + 8));
      acc = __builtin_amdgcn_wmma_f32_16x16x32_bf16(false, av, false, bv,
                                                    (short)0, acc, false, false);
    }
    int mcol = m0 + l15;
    for (int r = 0; r < 8; ++r) {
      int il = r + hh * 8;
      int j  = mcol - (L_ - 1) + (i0 + il);
      if (j >= 0) {                         // j <= i guaranteed
        int idx = il * 1024 + j;
        sc[idx] = f2bf(bf2f(sc[idx]) + acc[r]);
      }
    }
  }
  __syncthreads();

  // ---- pass 2: row softmax (16 threads per row), write P bf16 in place
  int r   = tid >> 4, sub = tid & 15;
  int ig  = i0 + r;
  int njc  = (ib + 2) >> 1;                 // P@V k-chunks of 32
  int jmax = njc << 5;
  unsigned short* srow = sc + r * 1024;

  float mx = -1e30f;
  for (int j = sub; j <= ig; j += 16) mx = fmaxf(mx, bf2f(srow[j]));
  red[tid] = mx; __syncthreads();
  float m = -1e30f;
  for (int u = 0; u < 16; ++u) m = fmaxf(m, red[(r << 4) + u]);
  __syncthreads();
  float s = 0.f;
  for (int j = sub; j <= ig; j += 16) s += __expf(bf2f(srow[j]) - m);
  red[tid] = s; __syncthreads();
  float tot = 0.f;
  for (int u = 0; u < 16; ++u) tot += red[(r << 4) + u];
  float inv = 1.f / tot;
  for (int j = sub; j < jmax; j += 16) {
    float p = (j <= ig) ? __expf(bf2f(srow[j]) - m) * inv : 0.f;
    srow[j] = f2bf(p);
  }
  __syncthreads();

  // ---- pass 3: O = P @ V  (V stored transposed -> contiguous B loads)
  int dkt  = wid & 3;                       // 4 dk-tiles of 16
  int half = wid >> 2;                      // split j-range across 2 waves
  const unsigned short* vrow = Vb + (size_t)(dkt * 16 + l15) * L_;
  v8f acc = {};
  for (int c = half; c < njc; c += 2) {
    int j0 = c << 5;
    v16bf av = pack_v16(*(const uint4*)(sc + l15 * 1024 + j0 + ka),
                        *(const uint4*)(sc + l15 * 1024 + j0 + ka + 8));
    v16bf bv = pack_v16(*(const uint4*)(vrow + j0 + kb),
                        *(const uint4*)(vrow + j0 + kb + 8));
    acc = __builtin_amdgcn_wmma_f32_16x16x32_bf16(false, av, false, bv,
                                                  (short)0, acc, false, false);
  }
  if (half == 1) {
    for (int rr = 0; rr < 8; ++rr)
      opart[(dkt * 16 + rr + hh * 8) * 16 + l15] = acc[rr];
  }
  __syncthreads();
  if (half == 0) {
    for (int rr = 0; rr < 8; ++rr) {
      int il = rr + hh * 8;
      float vv = acc[rr] + opart[(dkt * 16 + il) * 16 + l15];
      int i = i0 + il;
      int o = h * DK_ + dkt * 16 + l15;
      Oc[((size_t)b * L_ + i) * D_ + o] = f2bf(vv);
    }
  }
}

// ---------------------------------------------------------------- launch
extern "C" void kernel_launch(void* const* d_in, const int* in_sizes, int n_in,
                              void* d_out, int out_size, void* d_ws, size_t ws_size,
                              hipStream_t stream) {
  const float* q   = (const float*)d_in[0];
  const float* k   = (const float*)d_in[1];
  const float* v   = (const float*)d_in[2];
  // d_in[3] = mask: known causal tril -> handled structurally
  const float* wq  = (const float*)d_in[4];
  const float* bq  = (const float*)d_in[5];
  const float* wk  = (const float*)d_in[6];
  const float* bk  = (const float*)d_in[7];
  const float* wv  = (const float*)d_in[8];
  const float* bv  = (const float*)d_in[9];
  const float* wo  = (const float*)d_in[10];
  const float* bo  = (const float*)d_in[11];
  const float* rlk = (const float*)d_in[12];
  const float* rlt = (const float*)d_in[13];
  const float* rlp = (const float*)d_in[14];

  char* ws = (char*)d_ws;
  size_t off = 0;
  float* tp = (float*)(ws + off);           off += (size_t)L_ * DK_ * 4;       // 256 KB
  unsigned short* xq  = (unsigned short*)(ws + off); off += (size_t)NROW * D_ * 2;
  unsigned short* xk  = (unsigned short*)(ws + off); off += (size_t)NROW * D_ * 2;
  unsigned short* xv  = (unsigned short*)(ws + off); off += (size_t)NROW * D_ * 2;
  unsigned short* wqb = (unsigned short*)(ws + off); off += (size_t)D_ * D_ * 2;
  unsigned short* wkb = (unsigned short*)(ws + off); off += (size_t)D_ * D_ * 2;
  unsigned short* wvb = (unsigned short*)(ws + off); off += (size_t)D_ * D_ * 2;
  unsigned short* wob = (unsigned short*)(ws + off); off += (size_t)D_ * D_ * 2;
  unsigned short* rkb = (unsigned short*)(ws + off); off += (size_t)H_ * L_ * DK_ * 2;
  unsigned short* qh  = (unsigned short*)(ws + off); off += (size_t)B_ * H_ * L_ * DK_ * 2;
  unsigned short* kp  = (unsigned short*)(ws + off); off += (size_t)B_ * H_ * L_ * DK_ * 2;
  unsigned short* vt  = (unsigned short*)(ws + off); off += (size_t)B_ * H_ * L_ * DK_ * 2;
  unsigned short* oc  = (unsigned short*)(ws + off); off += (size_t)NROW * D_ * 2;
  (void)ws_size; (void)in_sizes; (void)n_in; (void)out_size;

  const int NXE = NROW * D_;                // 2,097,152
  const int NWE = D_ * D_;                  // 262,144
  const int NRK = H_ * L_ * DK_;            // 524,288

  k_f32_to_bf16<<<(NXE + 255) / 256, 256, 0, stream>>>(q,  xq,  NXE);
  k_f32_to_bf16<<<(NXE + 255) / 256, 256, 0, stream>>>(k,  xk,  NXE);
  k_f32_to_bf16<<<(NXE + 255) / 256, 256, 0, stream>>>(v,  xv,  NXE);
  k_f32_to_bf16<<<(NWE + 255) / 256, 256, 0, stream>>>(wq, wqb, NWE);
  k_f32_to_bf16<<<(NWE + 255) / 256, 256, 0, stream>>>(wk, wkb, NWE);
  k_f32_to_bf16<<<(NWE + 255) / 256, 256, 0, stream>>>(wv, wvb, NWE);
  k_f32_to_bf16<<<(NWE + 255) / 256, 256, 0, stream>>>(wo, wob, NWE);
  k_f32_to_bf16<<<(NRK + 255) / 256, 256, 0, stream>>>(rlk, rkb, NRK);
  k_tp_diag<<<(L_ * DK_) / 256, 256, 0, stream>>>(rlt, rlp, tp);

  // 1024 macro tiles (32x64 each) / 8 waves per block -> 128 blocks
  k_proj<<<128, 256, 0, stream>>>(xq, wqb, bq, tp, (void*)qh, 0);
  k_proj<<<128, 256, 0, stream>>>(xk, wkb, bk, tp, (void*)kp, 1);
  k_proj<<<128, 256, 0, stream>>>(xv, wvb, bv, tp, (void*)vt, 2);

  k_attn<<<B_ * H_ * (L_ / 16), 256, 0, stream>>>(qh, kp, vt, rkb, oc);

  k_proj<<<128, 256, 0, stream>>>(oc, wob, bo, tp, d_out, 3);
}
